// Rnet_34548716929428
// MI455X (gfx1250) — matrix operand order repaired and checked by reading, not compiled, not run on hardware
//
#include <hip/hip_runtime.h>
#include <math.h>
#include <stdint.h>

typedef _Float16 f16;
typedef __attribute__((ext_vector_type(16))) _Float16 v16h;
typedef __attribute__((ext_vector_type(8)))  _Float16 v8h;
typedef __attribute__((ext_vector_type(8)))  float    v8f;
typedef unsigned int v4u __attribute__((ext_vector_type(4)));
typedef int          v4i __attribute__((ext_vector_type(4)));
typedef int          v8i __attribute__((ext_vector_type(8)));

#define NBOX  4096
#define IMG_H 1080
#define IMG_W 1920

#if __has_builtin(__builtin_amdgcn_tensor_load_to_lds) && __has_builtin(__builtin_amdgcn_s_wait_tensorcnt)
#define HAVE_TDM 1
#else
#define HAVE_TDM 0
#endif
#if __has_include(<hip/amd_detail/amd_gfx1250_TDM.h>)
#define TDM_6ARG 1
#else
#define TDM_6ARG 0
#endif

// ---------------- workspace layout (bytes, all 16B aligned) ----------------
static constexpr size_t OFF_CROPS = 0;                      // 4096*1728*2
static constexpr size_t OFF_P1    = OFF_CROPS + 14155776;   // 4096*3388*2
static constexpr size_t OFF_C2    = OFF_P1    + 27754496;   // 4096*3888*2
static constexpr size_t OFF_P2    = OFF_C2    + 31850496;   // 4096*768*2
static constexpr size_t OFF_C3    = OFF_P2    + 6291456;    // 4096*576*2
static constexpr size_t OFF_L4    = OFF_C3    + 4718592;    // 4096*128*2
static constexpr size_t OFF_WB2   = OFF_L4    + 1048576;    // 48*256*2
static constexpr size_t OFF_WB3   = OFF_WB2   + 24576;      // 64*192*2
static constexpr size_t OFF_WB4   = OFF_WB3   + 24576;      // 128*576*2
static constexpr size_t OFF_SC    = OFF_WB4   + 147456;     // 4096*4   (f32 scores)
static constexpr size_t OFF_BX    = OFF_SC    + 16384;      // 4096*16  (f32 boxes)

__device__ __forceinline__ int imin(int a, int b) { return a < b ? a : b; }
__device__ __forceinline__ int imax(int a, int b) { return a > b ? a : b; }

// A-matrix fragment: 16 contiguous halves (per-lane K run, matches B)
__device__ __forceinline__ v16h ld_frag_a(const f16* p) {
    v8h lo = *(const v8h*)p;
    v8h hi = *(const v8h*)(p + 8);
    return __builtin_shufflevector(lo, hi, 0,1,2,3,4,5,6,7,8,9,10,11,12,13,14,15);
}
// B-matrix fragment: 16 contiguous halves (per-lane K run)
__device__ __forceinline__ v16h ld_frag_b(const f16* p) {
    v8h lo = *(const v8h*)p;
    v8h hi = *(const v8h*)(p + 8);
    return __builtin_shufflevector(lo, hi, 0,1,2,3,4,5,6,7,8,9,10,11,12,13,14,15);
}

// ---------------- TDM: 2D tile (rows x row_elems f16, stride == row_elems) -> LDS ----------------
__device__ __forceinline__ void tdm_load_2d(void* lds_dst, const void* gsrc,
                                            unsigned row_elems, unsigned rows) {
#if HAVE_TDM
    unsigned lds_addr = (unsigned)(uintptr_t)lds_dst;
    unsigned long long ga = (unsigned long long)(uintptr_t)gsrc;
    // D# group 0: count=1 | lds_addr | global_addr[56:0] | type=2
    v4u g0;
    g0.x = 1u;
    g0.y = lds_addr;
    g0.z = (unsigned)(ga & 0xffffffffull);
    g0.w = (unsigned)((ga >> 32) & 0x01ffffffull) | (2u << 30);
    // D# group 1: data_size=2B; tensor_dim0/1 = tile_dim0/1; dim0 stride = row_elems
    unsigned long long q0 = (1ull << 16)                                    // data_size=1 -> 2B
                          | ((unsigned long long)(row_elems & 0xffffu) << 48); // tensor_dim0[15:0]
    unsigned long long q1 = (unsigned long long)(row_elems >> 16)           // tensor_dim0[31:16]
                          | ((unsigned long long)rows << 16)                 // tensor_dim1
                          | ((unsigned long long)(row_elems & 0xffffu) << 48); // tile_dim0
    unsigned long long q2 = (unsigned long long)rows                         // tile_dim1 (tile_dim2=0)
                          | ((unsigned long long)row_elems << 32);           // tensor_dim0_stride[31:0]
    unsigned long long q3 = 0;                                               // stride[47:32]=0, dim1 stride=0
    v8i g1;
    g1[0] = (int)q0; g1[1] = (int)(q0 >> 32);
    g1[2] = (int)q1; g1[3] = (int)(q1 >> 32);
    g1[4] = (int)q2; g1[5] = (int)(q2 >> 32);
    g1[6] = (int)q3; g1[7] = (int)(q3 >> 32);
    v4i z4 = {0, 0, 0, 0};
#if TDM_6ARG
    v8i z8 = {0, 0, 0, 0, 0, 0, 0, 0};
    __builtin_amdgcn_tensor_load_to_lds(g0, g1, z4, z4, z8, 0);
#else
    __builtin_amdgcn_tensor_load_to_lds(g0, g1, z4, z4, 0);
#endif
#else
    (void)lds_dst; (void)gsrc; (void)row_elems; (void)rows;
#endif
}

// fallback contiguous global->LDS staging (tile rows contiguous)
__device__ __forceinline__ void stage_2d(f16* lds_dst, const f16* gsrc, int total_elems,
                                         int tid, int nthreads) {
    for (int i = tid * 8; i < total_elems; i += nthreads * 8)
        *(v8h*)(lds_dst + i) = *(const v8h*)(gsrc + i);
}

// ---------------- weight prep: f32 -> f16, pad K ----------------
__global__ void prep_w_kernel(const float* __restrict__ w2, const float* __restrict__ w3,
                              const float* __restrict__ w4,
                              f16* __restrict__ wb2, f16* __restrict__ wb3, f16* __restrict__ wb4) {
    const int tot = 12288 + 12288 + 73728;
    for (int idx = blockIdx.x * blockDim.x + threadIdx.x; idx < tot; idx += gridDim.x * blockDim.x) {
        if (idx < 12288) {                      // conv2: [48][252] -> [48][256]
            int oc = idx >> 8, k = idx & 255;
            wb2[idx] = (f16)(k < 252 ? w2[oc * 252 + k] : 0.f);
        } else if (idx < 24576) {               // conv3: [64][192]
            wb3[idx - 12288] = (f16)w3[idx - 12288];
        } else {                                // lin4: [128][576]
            wb4[idx - 24576] = (f16)w4[idx - 24576];
        }
    }
}

// ---------------- crop + normalize: one block per box ----------------
__global__ __launch_bounds__(256)
void crop_kernel(const float* __restrict__ image, const float* __restrict__ bb,
                 f16* __restrict__ crops) {
    const int b   = blockIdx.x;
    const int tid = threadIdx.x;
    __shared__ int ixs[24], iys[24];
    int x1 = imin(imax((int)bb[b*4+0], 0), IMG_W);
    int y1 = imin(imax((int)bb[b*4+1], 0), IMG_H);
    int x2 = imin(imax((int)bb[b*4+2], 0), IMG_W);
    int y2 = imin(imax((int)bb[b*4+3], 0), IMG_H);
    if (tid < 24) {
        ixs[tid] = imin(imax(x1 + (tid * (x2 - x1)) / 24, 0), IMG_W - 1);
        iys[tid] = imin(imax(y1 + (tid * (y2 - y1)) / 24, 0), IMG_H - 1);
    }
    __syncthreads();
    for (int idx = tid; idx < 3 * 576; idx += 256) {
        int c = idx / 576, rem = idx % 576, jy = rem / 24, jx = rem % 24;
        float v = image[((size_t)iys[jy] * IMG_W + ixs[jx]) * 3 + c];
        crops[(size_t)b * 1728 + idx] = (f16)((v - 127.5f) * 0.0078125f);
    }
}

// ---------------- conv1 + PReLU + maxpool(3,2,p1): one block per box ----------------
__global__ __launch_bounds__(256)
void conv1_kernel(const f16* __restrict__ crops, const float* __restrict__ w1,
                  const float* __restrict__ b1, const float* __restrict__ a1,
                  f16* __restrict__ p1) {
    const int b = blockIdx.x, tid = threadIdx.x;
    __shared__ __align__(16) f16 sc[1728];        // input crop
    __shared__ __align__(16) f16 sv[28 * 484];    // conv+prelu out 28x22x22
    __shared__ float sw[28 * 27];
    __shared__ float sb[28], sa[28];
    for (int i = tid; i < 1728; i += 256) sc[i] = crops[(size_t)b * 1728 + i];
    for (int i = tid; i < 756;  i += 256) sw[i] = w1[i];
    if (tid < 28) { sb[tid] = b1[tid]; sa[tid] = a1[tid]; }
    __syncthreads();
    for (int idx = tid; idx < 28 * 484; idx += 256) {
        int oc = idx / 484, rem = idx % 484, oh = rem / 22, ow = rem % 22;
        float s = sb[oc];
        const float* wp = &sw[oc * 27];
        #pragma unroll
        for (int ic = 0; ic < 3; ++ic)
            #pragma unroll
            for (int kh = 0; kh < 3; ++kh)
                #pragma unroll
                for (int kw = 0; kw < 3; ++kw)
                    s += (float)sc[ic * 576 + (oh + kh) * 24 + (ow + kw)] * wp[ic * 9 + kh * 3 + kw];
        s = s > 0.f ? s : sa[oc] * s;
        sv[idx] = (f16)s;
    }
    __syncthreads();
    for (int idx = tid; idx < 28 * 121; idx += 256) {
        int oc = idx / 121, rem = idx % 121, ph = rem / 11, pw = rem % 11;
        float m = -3.4e38f;
        for (int r = 2 * ph - 1; r <= 2 * ph + 1; ++r) {
            if (r < 0 || r >= 22) continue;
            for (int c = 2 * pw - 1; c <= 2 * pw + 1; ++c) {
                if (c < 0 || c >= 22) continue;
                m = fmaxf(m, (float)sv[oc * 484 + r * 22 + c]);
            }
        }
        p1[(size_t)b * 3388 + idx] = (f16)m;
    }
}

// ---------------- conv2 via WMMA implicit GEMM ----------------
// M = 4096 boxes, K = 252 (pad 256), N = 48. 3 waves/block (one N-tile each),
// shared im2col A patch + TDM-loaded B tile. grid = (256 M-tiles, 81 positions)
__global__ __launch_bounds__(96)
void conv2_wmma(const f16* __restrict__ p1, const f16* __restrict__ wb2,
                const float* __restrict__ bias, const float* __restrict__ alpha,
                f16* __restrict__ c2) {
    const int bx = blockIdx.x, pos = blockIdx.y;
    const int oh = pos / 9, ow = pos % 9;
    const int tid = threadIdx.x, wave = tid >> 5, lane = tid & 31;
    const int cbase = wave * 16, hs = lane >> 4, ln = lane & 15;
    __shared__ __align__(16) f16 sA[16 * 256];    // im2col patch for 16 boxes
    __shared__ __align__(16) f16 sB[48 * 256];    // full conv2 weight tile
#if HAVE_TDM
    if (wave == 0) tdm_load_2d(sB, wb2, 256, 48);
#else
    stage_2d(sB, wb2, 48 * 256, tid, 96);
#endif
    for (int idx = tid; idx < 16 * 256; idx += 96) {
        int mb = idx >> 8, k = idx & 255;
        f16 v = (f16)0.f;
        if (k < 252) {
            int ic = k / 9, r = (k % 9) / 3, c = k % 3;
            v = p1[(size_t)(bx * 16 + mb) * 3388 + ic * 121 + (oh + r) * 11 + (ow + c)];
        }
        sA[mb * 256 + k] = v;
    }
#if HAVE_TDM
    if (wave == 0) __builtin_amdgcn_s_wait_tensorcnt(0);
#endif
    __syncthreads();
    v8f acc = {};
    #pragma unroll
    for (int ks = 0; ks < 256; ks += 32) {
        v16h a  = ld_frag_a(&sA[ln * 256 + ks + hs * 16]);
        v16h bm = ld_frag_b(&sB[(cbase + ln) * 256 + ks + hs * 16]);
        acc = __builtin_amdgcn_wmma_f32_16x16x32_f16(false, a, false, bm, (short)0, acc, false, false);
    }
    const int n = cbase + ln;
    const float bs = bias[n], al = alpha[n];
    #pragma unroll
    for (int v = 0; v < 8; ++v) {
        int m = bx * 16 + v + hs * 8;
        float y = acc[v] + bs;
        y = y > 0.f ? y : al * y;
        c2[(size_t)m * 3888 + n * 81 + pos] = (f16)y;
    }
}

// ---------------- maxpool(3,2,p0) on conv2 output: 9x9 -> 4x4 ----------------
__global__ void pool2_kernel(const f16* __restrict__ c2, f16* __restrict__ p2) {
    const int total = NBOX * 48 * 16;
    for (int idx = blockIdx.x * blockDim.x + threadIdx.x; idx < total; idx += gridDim.x * blockDim.x) {
        int b = idx / 768, rem = idx % 768, oc = rem / 16, r2 = rem % 16, ph = r2 / 4, pw = r2 % 4;
        float m = -3.4e38f;
        for (int r = 2 * ph; r < 2 * ph + 3; ++r)
            for (int c = 2 * pw; c < 2 * pw + 3; ++c)
                m = fmaxf(m, (float)c2[(size_t)b * 3888 + oc * 81 + r * 9 + c]);
        p2[idx] = (f16)m;
    }
}

// ---------------- conv3 via WMMA implicit GEMM ----------------
// K = 192, N = 64. 4 waves/block. grid = (256, 9)
__global__ __launch_bounds__(128)
void conv3_wmma(const f16* __restrict__ p2, const f16* __restrict__ wb3,
                const float* __restrict__ bias, const float* __restrict__ alpha,
                f16* __restrict__ c3) {
    const int bx = blockIdx.x, pos = blockIdx.y;
    const int oh = pos / 3, ow = pos % 3;
    const int tid = threadIdx.x, wave = tid >> 5, lane = tid & 31;
    const int cbase = wave * 16, hs = lane >> 4, ln = lane & 15;
    __shared__ __align__(16) f16 sA[16 * 192];
    __shared__ __align__(16) f16 sB[64 * 192];
#if HAVE_TDM
    if (wave == 0) tdm_load_2d(sB, wb3, 192, 64);
#else
    stage_2d(sB, wb3, 64 * 192, tid, 128);
#endif
    for (int idx = tid; idx < 16 * 192; idx += 128) {
        int mb = idx / 192, k = idx % 192;
        int ic = k >> 2, kh = (k & 3) >> 1, kw = k & 1;
        sA[mb * 192 + k] = p2[(size_t)(bx * 16 + mb) * 768 + ic * 16 + (oh + kh) * 4 + (ow + kw)];
    }
#if HAVE_TDM
    if (wave == 0) __builtin_amdgcn_s_wait_tensorcnt(0);
#endif
    __syncthreads();
    v8f acc = {};
    #pragma unroll
    for (int ks = 0; ks < 192; ks += 32) {
        v16h a  = ld_frag_a(&sA[ln * 192 + ks + hs * 16]);
        v16h bm = ld_frag_b(&sB[(cbase + ln) * 192 + ks + hs * 16]);
        acc = __builtin_amdgcn_wmma_f32_16x16x32_f16(false, a, false, bm, (short)0, acc, false, false);
    }
    const int n = cbase + ln;
    const float bs = bias[n], al = alpha[n];
    #pragma unroll
    for (int v = 0; v < 8; ++v) {
        int m = bx * 16 + v + hs * 8;
        float y = acc[v] + bs;
        y = y > 0.f ? y : al * y;
        // flattened feature index = oc*9 + oh*3 + ow (matches reference reshape)
        c3[(size_t)m * 576 + n * 9 + pos] = (f16)y;
    }
}

// ---------------- lin4 (4096x576 @ 576x128) via WMMA, fused PReLU ----------------
// 8 waves/block (one N-tile each) sharing a TDM-loaded A tile. grid = 256
__global__ __launch_bounds__(256)
void lin4_wmma(const f16* __restrict__ c3, const f16* __restrict__ wb4,
               const float* __restrict__ bias, const float* __restrict__ alpha,
               f16* __restrict__ l4) {
    const int bx = blockIdx.x;
    const int tid = threadIdx.x, wave = tid >> 5, lane = tid & 31;
    const int cbase = wave * 16, hs = lane >> 4, ln = lane & 15;
    __shared__ __align__(16) f16 sA[16 * 576];
    const f16* ga = c3 + (size_t)bx * 16 * 576;
#if HAVE_TDM
    if (wave == 0) {
        tdm_load_2d(sA, ga, 576, 16);
        __builtin_amdgcn_s_wait_tensorcnt(0);
    }
#else
    stage_2d(sA, ga, 16 * 576, tid, 256);
#endif
    __syncthreads();
    const f16* pb = wb4 + (size_t)(cbase + ln) * 576 + hs * 16;
    v8f acc = {};
    #pragma unroll
    for (int ks = 0; ks < 576; ks += 32) {
        __builtin_prefetch(pb + ks + 64, 0, 1);
        v16h a  = ld_frag_a(&sA[ln * 576 + ks + hs * 16]);
        v16h bm = ld_frag_b(pb + ks);
        acc = __builtin_amdgcn_wmma_f32_16x16x32_f16(false, a, false, bm, (short)0, acc, false, false);
    }
    const int n = cbase + ln;
    const float bs = bias[n], al = alpha[n];
    #pragma unroll
    for (int v = 0; v < 8; ++v) {
        int m = bx * 16 + v + hs * 8;
        float y = acc[v] + bs;
        y = y > 0.f ? y : al * y;
        l4[(size_t)m * 128 + n] = (f16)y;
    }
}

// ---------------- heads: cls softmax + box regression ----------------
__global__ void head_kernel(const f16* __restrict__ l4,
                            const float* __restrict__ w5a, const float* __restrict__ b5a,
                            const float* __restrict__ w5b, const float* __restrict__ b5b,
                            const float* __restrict__ bb,
                            float* __restrict__ boxes, float* __restrict__ scores) {
    int b = blockIdx.x * blockDim.x + threadIdx.x;
    if (b >= NBOX) return;
    float c0 = b5a[0], c1 = b5a[1];
    float r0 = b5b[0], r1 = b5b[1], r2 = b5b[2], r3 = b5b[3];
    for (int k = 0; k < 128; ++k) {
        float x = (float)l4[(size_t)b * 128 + k];
        c0 += x * w5a[k];        c1 += x * w5a[128 + k];
        r0 += x * w5b[k];        r1 += x * w5b[128 + k];
        r2 += x * w5b[256 + k];  r3 += x * w5b[384 + k];
    }
    float mx = fmaxf(c0, c1);
    float e0 = expf(c0 - mx), e1 = expf(c1 - mx);
    scores[b] = e1 / (e0 + e1);
    float x1 = bb[b*4], y1 = bb[b*4+1], x2 = bb[b*4+2], y2 = bb[b*4+3];
    float w = x2 - x1, h = y2 - y1;
    boxes[b*4+0] = x1 + r0 * w;
    boxes[b*4+1] = y1 + r1 * h;
    boxes[b*4+2] = x2 + r2 * w;
    boxes[b*4+3] = y2 + r3 * h;
}

// ---------------- NMS: bitonic sort + sequential suppression, single workgroup ----------------
__global__ __launch_bounds__(1024)
void nms_kernel(const float* __restrict__ boxes, const float* __restrict__ scores,
                float* __restrict__ out) {
    __shared__ float ss[NBOX];
    __shared__ int   si[NBOX];
    __shared__ unsigned char supp[NBOX];
    const int tid = threadIdx.x;
    for (int i = tid; i < NBOX; i += 1024) { ss[i] = scores[i]; si[i] = i; supp[i] = 0; }
    __syncthreads();
    // bitonic sort descending by score
    for (int k = 2; k <= NBOX; k <<= 1) {
        for (int j = k >> 1; j > 0; j >>= 1) {
            for (int i = tid; i < NBOX; i += 1024) {
                int l = i ^ j;
                if (l > i) {
                    bool desc = ((i & k) == 0);
                    float a = ss[i], b = ss[l];
                    bool sw = desc ? (a < b) : (a > b);
                    if (sw) {
                        ss[i] = b; ss[l] = a;
                        int t = si[i]; si[i] = si[l]; si[l] = t;
                    }
                }
            }
            __syncthreads();
        }
    }
    // sequential suppression sweep (matches reference fori_loop)
    for (int i = 0; i < NBOX; ++i) {
        bool alive = (ss[i] >= 0.7f) && (supp[i] == 0);   // uniform across block
        if (alive) {
            const float* bi = boxes + (size_t)si[i] * 4;
            float ax1 = bi[0], ay1 = bi[1], ax2 = bi[2], ay2 = bi[3];
            float aarea = fmaxf(ax2 - ax1, 0.f) * fmaxf(ay2 - ay1, 0.f);
            for (int jj = i + 1 + tid; jj < NBOX; jj += 1024) {
                const float* bj = boxes + (size_t)si[jj] * 4;
                float bx1 = bj[0], by1 = bj[1], bx2 = bj[2], by2 = bj[3];
                float ix1 = fmaxf(ax1, bx1), iy1 = fmaxf(ay1, by1);
                float ix2 = fminf(ax2, bx2), iy2 = fminf(ay2, by2);
                float inter = fmaxf(ix2 - ix1, 0.f) * fmaxf(iy2 - iy1, 0.f);
                float barea = fmaxf(bx2 - bx1, 0.f) * fmaxf(by2 - by1, 0.f);
                float iou = inter / fmaxf(aarea + barea - inter, 1e-12f);
                if (iou > 0.5f) supp[jj] = 1;
            }
        }
        __syncthreads();
    }
    for (int i = tid; i < NBOX; i += 1024) {
        int orig = si[i];
        bool keep = (ss[i] >= 0.7f) && (supp[i] == 0);
        #pragma unroll
        for (int c = 0; c < 4; ++c)
            out[(size_t)orig * 4 + c] = keep ? boxes[(size_t)orig * 4 + c] : 0.f;
    }
}

// ---------------- host launch ----------------
extern "C" void kernel_launch(void* const* d_in, const int* in_sizes, int n_in,
                              void* d_out, int out_size, void* d_ws, size_t ws_size,
                              hipStream_t stream) {
    (void)in_sizes; (void)n_in; (void)out_size; (void)ws_size;
    const float* image   = (const float*)d_in[0];
    const float* bboxes  = (const float*)d_in[1];
    const float* conv1_w = (const float*)d_in[2];
    const float* conv1_b = (const float*)d_in[3];
    const float* prelu1  = (const float*)d_in[4];
    const float* conv2_w = (const float*)d_in[5];
    const float* conv2_b = (const float*)d_in[6];
    const float* prelu2  = (const float*)d_in[7];
    const float* conv3_w = (const float*)d_in[8];
    const float* conv3_b = (const float*)d_in[9];
    const float* prelu3  = (const float*)d_in[10];
    const float* lin4_w  = (const float*)d_in[11];
    const float* lin4_b  = (const float*)d_in[12];
    const float* prelu4  = (const float*)d_in[13];
    const float* l5a_w   = (const float*)d_in[14];
    const float* l5a_b   = (const float*)d_in[15];
    const float* l5b_w   = (const float*)d_in[16];
    const float* l5b_b   = (const float*)d_in[17];

    char* ws = (char*)d_ws;
    f16*   crops  = (f16*)(ws + OFF_CROPS);
    f16*   p1     = (f16*)(ws + OFF_P1);
    f16*   c2     = (f16*)(ws + OFF_C2);
    f16*   p2     = (f16*)(ws + OFF_P2);
    f16*   c3     = (f16*)(ws + OFF_C3);
    f16*   l4     = (f16*)(ws + OFF_L4);
    f16*   wb2    = (f16*)(ws + OFF_WB2);
    f16*   wb3    = (f16*)(ws + OFF_WB3);
    f16*   wb4    = (f16*)(ws + OFF_WB4);
    float* scr    = (float*)(ws + OFF_SC);
    float* bxs    = (float*)(ws + OFF_BX);
    float* out    = (float*)d_out;

    prep_w_kernel<<<96, 256, 0, stream>>>(conv2_w, conv3_w, lin4_w, wb2, wb3, wb4);
    crop_kernel<<<NBOX, 256, 0, stream>>>(image, bboxes, crops);
    conv1_kernel<<<NBOX, 256, 0, stream>>>(crops, conv1_w, conv1_b, prelu1, p1);
    conv2_wmma<<<dim3(NBOX/16, 81), 96, 0, stream>>>(p1, wb2, conv2_b, prelu2, c2);
    pool2_kernel<<<2048, 256, 0, stream>>>(c2, p2);
    conv3_wmma<<<dim3(NBOX/16, 9), 128, 0, stream>>>(p2, wb3, conv3_b, prelu3, c3);
    lin4_wmma<<<NBOX/16, 256, 0, stream>>>(c3, wb4, lin4_b, prelu4, l4);
    head_kernel<<<(NBOX + 255) / 256, 256, 0, stream>>>(l4, l5a_w, l5a_b, l5b_w, l5b_b,
                                                        bboxes, bxs, scr);
    nms_kernel<<<1, 1024, 0, stream>>>(bxs, scr, out);
}
